// NeighborEmbedding_14697378087510
// MI455X (gfx1250) — matrix operand order repaired and checked by reading, not compiled, not run on hardware
//
#include <hip/hip_runtime.h>

typedef __attribute__((ext_vector_type(16))) __bf16 v16bf;
typedef __attribute__((ext_vector_type(8)))  float  v8f;

#define HIDDEN 128
#define NRBF   64
#define PI_F   3.14159265358979f

__device__ __forceinline__ __bf16 tobf(float x) { return (__bf16)x; }

// ---------------------------------------------------------------- zero agg
__global__ void ne_zero_kernel(float* __restrict__ p, long n) {
  long i = (long)blockIdx.x * blockDim.x + threadIdx.x;
  const long stride = (long)gridDim.x * blockDim.x;
  for (; i < n; i += stride) p[i] = 0.0f;
}

// ---------------------------------------------------------------- edge kernel
// 128 threads = 4 waves; each wave owns 16 edges.
// W = (edge_attr @ Wd + bd) * (cutoff*mask);  xn = emb[z[row]] * W;
// scatter-add xn into agg[col] with native f32 global atomics.
__global__ void __launch_bounds__(128)
ne_edge_kernel(const int* __restrict__ ei, const float* __restrict__ ew,
               const float* __restrict__ ea, const int* __restrict__ z,
               const float* __restrict__ emb, const float* __restrict__ Wd,
               const float* __restrict__ bd, float* __restrict__ agg,
               int E) {
  // Wd staged as bf16 in per-lane WMMA-B layout: [ks(2)][nt(8)][lane(32)][16]
  __shared__ __attribute__((aligned(32))) __bf16 ldsB[2 * 8 * 32 * 16]; // 16 KB
  __shared__ float sS[4][16];
  __shared__ int   sZr[4][16];
  __shared__ int   sCol[4][16];

  const int tid = threadIdx.x;

  // Stage Wd (f32 row-major [64][128]) -> bf16 B-layout chunks of 16.
  for (int c = tid; c < 2 * 8 * 32; c += 128) {
    const int lane  = c & 31;
    const int nt    = (c >> 5) & 7;
    const int ks    = c >> 8;
    const int kBase = ks * 32 + (lane >> 4) * 16;  // B: lanes 16-31 hold K+16
    const int n     = nt * 16 + (lane & 15);
    __bf16* dst = &ldsB[c * 16];
#pragma unroll
    for (int i = 0; i < 16; ++i)
      dst[i] = tobf(Wd[(kBase + i) * HIDDEN + n]);
  }

  const int wave = tid >> 5;
  const int lane = tid & 31;
  const int m    = lane & 15;   // A row within tile / C column within tile
  const int hi   = lane >> 4;
  const long eBase = (long)blockIdx.x * 64 + (long)wave * 16;

  // Per-edge metadata: cutoff*mask scalar, z[row], col.
  if (lane < 16) {
    const long e = eBase + lane;
    float s = 0.0f; int zr = 0, cc = 0;
    if (e < E) {
      const int r = ei[e];
      cc = ei[(long)E + e];
      const float w = ew[e];
      const float C = 0.5f * (__cosf(w * (PI_F / 5.0f)) + 1.0f);
      s = (r != cc && w < 5.0f) ? C : 0.0f;
      zr = z[r];
    }
    sS[wave][lane]   = s;
    sZr[wave][lane]  = zr;
    sCol[wave][lane] = cc;
  }
  __syncthreads();

  // A tiles: 16x32 bf16 per K-step. Layout: lanes 0-15 hold K {0-15},
  // lanes 16-31 hold K {16-31} of row m (matches B staging).
  long e = eBase + m;
  if (e >= E) e = (long)E - 1;              // clamped; s==0 guards the scatter
  const float* arow = ea + e * NRBF;
  v16bf A[2];
#pragma unroll
  for (int ks = 0; ks < 2; ++ks) {
    const int s1 = ks * 32 + 16 * hi;
#pragma unroll
    for (int i = 0; i < 8; ++i) A[ks][i]     = tobf(arow[s1 + i]);
#pragma unroll
    for (int i = 0; i < 8; ++i) A[ks][8 + i] = tobf(arow[s1 + 8 + i]);
  }

  // All 8 N-tile accumulators (64 VGPRs), so the epilogue can go row-major.
  v8f acc[8];
#pragma unroll
  for (int nt = 0; nt < 8; ++nt) {
    v8f a = {};
    const v16bf B0 = *(const v16bf*)&ldsB[((0 * 8 + nt) * 32 + lane) * 16];
    const v16bf B1 = *(const v16bf*)&ldsB[((1 * 8 + nt) * 32 + lane) * 16];
    a = __builtin_amdgcn_wmma_f32_16x16x32_bf16(false, A[0], false, B0,
                                                (short)0, a, false, false);
    a = __builtin_amdgcn_wmma_f32_16x16x32_bf16(false, A[1], false, B1,
                                                (short)0, a, false, false);
    acc[nt] = a;
  }

  float bdv[8];
#pragma unroll
  for (int nt = 0; nt < 8; ++nt) bdv[nt] = bd[nt * 16 + m];

  // Epilogue: one edge (row mm) at a time; 8 gathers + 8 atomics share a
  // single base address with immediate offsets (+64B steps).
#pragma unroll
  for (int r = 0; r < 8; ++r) {
    const int mm = r + 8 * hi;              // C layout: row = r + 8*(lane>=16)
    const float s = sS[wave][mm];
    if (s != 0.0f) {
      const float* embp = emb + (long)sZr[wave][mm]  * HIDDEN + m;
      float*       aggp = agg + (long)sCol[wave][mm] * HIDDEN + m;
#pragma unroll
      for (int nt = 0; nt < 8; ++nt) {
        const float wv = (acc[nt][r] + bdv[nt]) * s;
        __hip_atomic_fetch_add(aggp + nt * 16, embp[nt * 16] * wv,
                               __ATOMIC_RELAXED, __HIP_MEMORY_SCOPE_AGENT);
      }
    }
  }
}

// ---------------------------------------------------------------- node kernel
// out = concat(x, agg) @ Wc + bc.  256 threads = 8 waves; 16 rows per wave.
__global__ void __launch_bounds__(256)
ne_node_kernel(const float* __restrict__ x, const float* __restrict__ agg,
               const float* __restrict__ Wc, const float* __restrict__ bc,
               float* __restrict__ out, int N) {
  // Wc staged as bf16 B-layout: [ks(8)][nt(8)][lane(32)][16] = 64 KB
  __shared__ __attribute__((aligned(32))) __bf16 ldsB[8 * 8 * 32 * 16];
  const int tid = threadIdx.x;

  for (int c = tid; c < 8 * 8 * 32; c += 256) {
    const int lane  = c & 31;
    const int nt    = (c >> 5) & 7;
    const int ks    = c >> 8;
    const int kBase = ks * 32 + (lane >> 4) * 16;
    const int n     = nt * 16 + (lane & 15);
    __bf16* dst = &ldsB[c * 16];
#pragma unroll
    for (int i = 0; i < 16; ++i)
      dst[i] = tobf(Wc[(kBase + i) * HIDDEN + n]);
  }
  __syncthreads();

  const int wave = tid >> 5;
  const int lane = tid & 31;
  const int m    = lane & 15;
  const int hi   = lane >> 4;
  const long rowBase = (long)blockIdx.x * 128 + (long)wave * 16;
  long row = rowBase + m;
  if (row >= N) row = (long)N - 1;

  // A from concat(x,agg): K<128 -> x, K>=128 -> agg.  8-element groups are
  // 8-aligned so they never straddle the boundary.
  v16bf A[8];
#pragma unroll
  for (int ks = 0; ks < 8; ++ks) {
#pragma unroll
    for (int g = 0; g < 2; ++g) {
      const int kk = ks * 32 + 16 * hi + 8 * g;
      const float* src = (kk < HIDDEN) ? (x + row * HIDDEN + kk)
                                       : (agg + row * HIDDEN + (kk - HIDDEN));
#pragma unroll
      for (int i = 0; i < 8; ++i) A[ks][8 * g + i] = tobf(src[i]);
    }
  }

  v8f acc[8];
#pragma unroll
  for (int nt = 0; nt < 8; ++nt) {
    v8f a = {};
#pragma unroll
    for (int ks = 0; ks < 8; ++ks) {
      const v16bf B = *(const v16bf*)&ldsB[((ks * 8 + nt) * 32 + lane) * 16];
      a = __builtin_amdgcn_wmma_f32_16x16x32_bf16(false, A[ks], false, B,
                                                  (short)0, a, false, false);
    }
    acc[nt] = a;
  }

  float bcv[8];
#pragma unroll
  for (int nt = 0; nt < 8; ++nt) bcv[nt] = bc[nt * 16 + m];

  // Row-major epilogue: one bounds check + one base address per output row;
  // 8 stores with immediate offsets.
#pragma unroll
  for (int r = 0; r < 8; ++r) {
    const long rr = rowBase + r + 8 * hi;
    if (rr < N) {
      float* op = out + rr * HIDDEN + m;
#pragma unroll
      for (int nt = 0; nt < 8; ++nt) op[nt * 16] = acc[nt][r] + bcv[nt];
    }
  }
}

// ---------------------------------------------------------------- launch
extern "C" void kernel_launch(void* const* d_in, const int* in_sizes, int n_in,
                              void* d_out, int out_size, void* d_ws, size_t ws_size,
                              hipStream_t stream) {
  const int*   z   = (const int*)  d_in[0];
  const float* x   = (const float*)d_in[1];
  const int*   ei  = (const int*)  d_in[2];
  const float* ew  = (const float*)d_in[3];
  const float* ea  = (const float*)d_in[4];
  const float* emb = (const float*)d_in[5];
  const float* Wd  = (const float*)d_in[6];
  const float* bd  = (const float*)d_in[7];
  const float* Wc  = (const float*)d_in[8];
  const float* bc  = (const float*)d_in[9];
  float* out = (float*)d_out;

  const int N = in_sizes[0];        // z count = N_NODES
  const int E = in_sizes[3];        // edge_weight count = N_EDGES

  float* agg = (float*)d_ws;        // [N, HIDDEN] f32 scratch
  const long aggN = (long)N * HIDDEN;

  ne_zero_kernel<<<2048, 256, 0, stream>>>(agg, aggN);
  ne_edge_kernel<<<(E + 63) / 64, 128, 0, stream>>>(ei, ew, ea, z, emb, Wd, bd,
                                                    agg, E);
  ne_node_kernel<<<(N + 127) / 128, 256, 0, stream>>>(x, agg, Wc, bc, out, N);
}
